// source_model_19396072308786
// MI455X (gfx1250) — matrix operand order repaired and checked by reading, not compiled, not run on hardware
//
#include <hip/hip_runtime.h>
#include <hip/hip_bf16.h>

// Problem constants (from the reference)
#define VOCAB 30
#define TCH   8192      // output channels (argmax dimension, GEMM N)
#define KW    5         // conv width
#define SEQ   512
#define BATCH 32
#define ROWS  (BATCH * SEQ)   // 16384 GEMM M rows
#define KPAD  160             // V*K=150 padded to 5 k-steps of 32 (row150 = bias)
#define KTOT  320             // hi + lo split-precision halves
#define NTILES (TCH / 16)     // 512

typedef __attribute__((ext_vector_type(16))) _Float16 v16h;
typedef __attribute__((ext_vector_type(8)))  _Float16 v8h;
typedef __attribute__((ext_vector_type(8)))  float    v8f;

// ---------------------------------------------------------------------------
// Prep 1: Bmat[t][0..159] = f16 hi of (w[t][kp] | bias | 0),
//         Bmat[t][160..319] = f16 lo residual  (split-precision f32 recovery)
// w layout (T,V,K) flat => element (t, v, k) at t*150 + (v*5 + k) = t*150 + kp
// ---------------------------------------------------------------------------
__global__ void prep_b_kernel(const float* __restrict__ w,
                              const float* __restrict__ bias,
                              _Float16* __restrict__ Bmat) {
    int i = blockIdx.x * 256 + threadIdx.x;     // i = t*160 + kp
    if (i >= TCH * KPAD) return;
    int t  = i / KPAD;
    int kp = i - t * KPAD;
    float val = 0.0f;
    if (kp < VOCAB * KW)      val = w[t * (VOCAB * KW) + kp];
    else if (kp == VOCAB * KW) val = bias[t];
    _Float16 hi = (_Float16)val;
    _Float16 lo = (_Float16)(val - (float)hi);
    Bmat[(size_t)t * KTOT + kp]        = hi;
    Bmat[(size_t)t * KTOT + KPAD + kp] = lo;
}

// ---------------------------------------------------------------------------
// Prep 2: 5-hot A matrix, A[r][v*5+k] = 1 for v = s[b, l+k-2] (zero-padded
// conv borders), A[r][150] = 1 (bias column). r = b*512 + l.
// ---------------------------------------------------------------------------
__global__ void prep_a_kernel(const int* __restrict__ s,
                              _Float16* __restrict__ Amat) {
    int r = blockIdx.x * 256 + threadIdx.x;
    if (r >= ROWS) return;
    _Float16* row = Amat + (size_t)r * KPAD;    // row base = 320 B, 16B aligned
    float4 z = make_float4(0.f, 0.f, 0.f, 0.f);
#pragma unroll
    for (int i = 0; i < KPAD * 2 / 16; ++i) ((float4*)row)[i] = z;
    int b = r >> 9;
    int l = r & (SEQ - 1);
    const int* sb = s + b * SEQ;
#pragma unroll
    for (int k = 0; k < KW; ++k) {
        int pos = l + k - (KW / 2);
        if (pos >= 0 && pos < SEQ) {
            int v = sb[pos];
            row[v * KW + k] = (_Float16)1.0f;
        }
    }
    row[VOCAB * KW] = (_Float16)1.0f;           // bias column
}

// ---------------------------------------------------------------------------
// GEMM + fused argmax.
// Block = 256 threads = 8 waves. Wave w owns M-tile rows [blk*128 + w*16, +16).
// A fragments (5 k-steps) pinned in VGPRs; sweep all 512 N-tiles with
// 10 chained v_wmma_f32_16x16x32_f16 (hi+lo passes) per tile, fused running
// argmax per lane, LDS u64 atomicMax reduction with JAX-compatible tie-break.
// ---------------------------------------------------------------------------
__global__ __launch_bounds__(256) void gemm_argmax_kernel(
        const _Float16* __restrict__ Amat,
        const _Float16* __restrict__ Bmat,
        float* __restrict__ out) {
    __shared__ unsigned long long red[128];
    const int tid = threadIdx.x;
    if (tid < 128) red[tid] = 0ull;
    __syncthreads();

    const int wave = tid >> 5;
    const int lane = tid & 31;
    const int g    = lane >> 4;       // lane group (0: lanes 0-15, 1: 16-31)
    const int ln   = lane & 15;
    const int mBase = blockIdx.x * 128 + wave * 16;

    // ---- Load A fragments (ISA 16-bit A 16x32 layout) ----
    // lane M = ln; half h -> K = kb + 16*(h/8) + (h%8) + 8*g
    v16h afrag[5];
    const _Float16* Arow = Amat + (size_t)(mBase + ln) * KPAD;
#pragma unroll
    for (int kk = 0; kk < 5; ++kk) {
        const int kb = kk * 32;
        v8h lo = *(const v8h*)(Arow + kb + 8 * g);
        v8h hi = *(const v8h*)(Arow + kb + 16 + 8 * g);
#pragma unroll
        for (int i = 0; i < 8; ++i) { afrag[kk][i] = lo[i]; afrag[kk][8 + i] = hi[i]; }
    }

    // ---- Running argmax state: C/D layout VGPR j -> (M = 8*g + j, N = ln) ----
    float mx[8];
    int   ix[8];
#pragma unroll
    for (int j = 0; j < 8; ++j) { mx[j] = -3.402823466e38f; ix[j] = 0; }

    for (int nt = 0; nt < NTILES; ++nt) {
        const int n = nt * 16 + ln;                 // this lane's channel column
        const _Float16* Brow = Bmat + (size_t)n * KTOT;
        v8f c = {};
#pragma unroll
        for (int kk = 0; kk < 10; ++kk) {           // kk<5: hi halves, kk>=5: lo
            const int kb = kk * 32;
            // ISA 16-bit B 32x16 layout: half h -> K = kb + 16*(h/8) + (h%8) + 8*g
            v8h blo = *(const v8h*)(Brow + kb + 8 * g);
            v8h bhi = *(const v8h*)(Brow + kb + 16 + 8 * g);
            v16h bf;
#pragma unroll
            for (int i = 0; i < 8; ++i) { bf[i] = blo[i]; bf[8 + i] = bhi[i]; }
            c = __builtin_amdgcn_wmma_f32_16x16x32_f16(
                    /*neg_a=*/false, afrag[kk % 5],
                    /*neg_b=*/false, bf,
                    /*c_mod=*/(short)0, c,
                    /*reuse_a=*/false, /*reuse_b=*/false);
        }
#pragma unroll
        for (int j = 0; j < 8; ++j) {
            if (c[j] > mx[j]) { mx[j] = c[j]; ix[j] = n; }
        }
    }

    // ---- Reduce: sortable key = (order-flipped f32 << 32) | ~idx ----
    // max key => max value, ties => smallest index (matches jnp.argmax).
#pragma unroll
    for (int j = 0; j < 8; ++j) {
        unsigned u = __float_as_uint(mx[j]);
        u = (u & 0x80000000u) ? ~u : (u | 0x80000000u);
        unsigned long long key =
            ((unsigned long long)u << 32) | (unsigned)(~ix[j]);
        int row = g * 8 + j;
        atomicMax(&red[wave * 16 + row], key);
    }
    __syncthreads();

    if (tid < 128) {
        unsigned long long k = red[tid];
        int best = (int)(~(unsigned)(k & 0xFFFFFFFFull));
        // slot tid = wave*16 + row  ->  global row = blk*128 + tid
        out[blockIdx.x * 128 + tid] = (float)best;
    }
}

// ---------------------------------------------------------------------------
// Launch
// ---------------------------------------------------------------------------
extern "C" void kernel_launch(void* const* d_in, const int* in_sizes, int n_in,
                              void* d_out, int out_size, void* d_ws, size_t ws_size,
                              hipStream_t stream) {
    const int*   s    = (const int*)d_in[0];     // (B, L) token ids
    const float* w    = (const float*)d_in[1];   // (T, V, K)
    const float* bias = (const float*)d_in[2];   // (T,)
    float* out = (float*)d_out;                  // (B, L) argmax indices as f32

    // Workspace layout: Bmat 8192*320*2 = 5,242,880 B ; Amat 16384*160*2 = 5,242,880 B
    _Float16* Bmat = (_Float16*)d_ws;
    _Float16* Amat = (_Float16*)((char*)d_ws + (size_t)TCH * KTOT * sizeof(_Float16));

    prep_b_kernel<<<(TCH * KPAD + 255) / 256, 256, 0, stream>>>(w, bias, Bmat);
    prep_a_kernel<<<(ROWS + 255) / 256, 256, 0, stream>>>(s, Amat);
    gemm_argmax_kernel<<<ROWS / 128, 256, 0, stream>>>(Amat, Bmat, out);
}
